// LLaMAAttention_27693949125388
// MI455X (gfx1250) — matrix operand order repaired and checked by reading, not compiled, not run on hardware
//
#include <hip/hip_runtime.h>

#define NH     32
#define NKV    8
#define HD     128
#define DMODEL 4096
#define KVD    1024

typedef __bf16 bf16_t;
typedef bf16_t v16bf __attribute__((ext_vector_type(16)));
typedef float  v8f   __attribute__((ext_vector_type(8)));
typedef int    v4i_t __attribute__((ext_vector_type(4)));

// ---------------- CDNA5 async global->LDS path (guarded) ----------------
#if defined(__has_builtin)
#if __has_builtin(__builtin_amdgcn_global_load_async_to_lds_b128) && \
    __has_builtin(__builtin_amdgcn_s_wait_asynccnt)
#define USE_ASYNC 1
#endif
#endif
#ifndef USE_ASYNC
#define USE_ASYNC 0
#endif

#if USE_ASYNC
typedef __attribute__((address_space(1))) v4i_t gas_v4i;
typedef __attribute__((address_space(3))) v4i_t las_v4i;
__device__ __forceinline__ void async_b128(const void* g, void* l) {
  __builtin_amdgcn_global_load_async_to_lds_b128((gas_v4i*)g, (las_v4i*)l, 0, 0);
}
__device__ __forceinline__ void async_wait0() {
  __builtin_amdgcn_s_wait_asynccnt(0);
}
#define STAGE16B(g, l) async_b128((g), (l))
#else
__device__ __forceinline__ void async_wait0() {}
#define STAGE16B(g, l) (*(uint4*)(l) = *(const uint4*)(g))
#endif

__device__ __forceinline__ bf16_t f2bf(float f) {
  unsigned u = __builtin_bit_cast(unsigned, f);
  unsigned r = u + 0x7FFFu + ((u >> 16) & 1u);   // round-to-nearest-even
  unsigned short h = (unsigned short)(r >> 16);
  return __builtin_bit_cast(bf16_t, h);
}

// ---------------------------------------------------------------- elementwise
__global__ void k_f32_to_bf16(const float* __restrict__ in,
                              bf16_t* __restrict__ out, long n) {
  long i = (long)blockIdx.x * blockDim.x + threadIdx.x;
  long stride = (long)gridDim.x * blockDim.x;
  for (; i < n; i += stride) out[i] = f2bf(in[i]);
}

// ------------------------------------------------------------------- RoPE (Q)
// in: f32 [T][H*128] -> out: bf16 [T][H*128]
__global__ void k_rope_to_bf16(const float* __restrict__ in,
                               bf16_t* __restrict__ out, int T, int H) {
  int id = blockIdx.x * blockDim.x + threadIdx.x;
  int total = T * H * (HD / 2);
  if (id >= total) return;
  int i = id % (HD / 2);
  int h = (id / (HD / 2)) % H;
  int t = id / (H * (HD / 2));
  float inv = __powf(10000.0f, -(float)i / 64.0f);
  float ang = (float)t * inv;
  float c = __cosf(ang), s = __sinf(ang);
  const float* row = in + (size_t)t * (H * HD) + h * HD;
  bf16_t* orow = out + (size_t)t * (H * HD) + h * HD;
  float x1 = row[i], x2 = row[i + 64];
  orow[i]      = f2bf(x1 * c - x2 * s);
  orow[i + 64] = f2bf(x2 * c + x1 * s);
}

// ------------------------------------------------------------------- RoPE (K)
// in: f32 [T][KVD] -> out: bf16 TRANSPOSED [NKV*HD][T]  (so the attention
// kernel can stage K^T tiles with contiguous 16B async loads)
__global__ void k_rope_kt(const float* __restrict__ in,
                          bf16_t* __restrict__ out, int T) {
  int id = blockIdx.x * blockDim.x + threadIdx.x;
  int total = T * NKV * (HD / 2);
  if (id >= total) return;
  int i = id % (HD / 2);
  int kvh = (id / (HD / 2)) % NKV;
  int t = id / (NKV * (HD / 2));
  float inv = __powf(10000.0f, -(float)i / 64.0f);
  float ang = (float)t * inv;
  float c = __cosf(ang), s = __sinf(ang);
  const float* row = in + (size_t)t * KVD + kvh * HD;
  float x1 = row[i], x2 = row[i + 64];
  out[((size_t)(kvh * HD + i)) * T + t]      = f2bf(x1 * c - x2 * s);
  out[((size_t)(kvh * HD + i + 64)) * T + t] = f2bf(x2 * c + x1 * s);
}

// ------------------------------------------------------------------- GEMM
// C[M,N](f32) = A[M,K](bf16,rm) * B[K,N](bf16,rm)
// 256 thr (8 waves). Tile BM=128 x BN=128, BK=32, double-buffered LDS,
// async global->LDS staging overlapped with WMMA.
#define BM 128
#define BN 128
#define BK 32

__global__ __launch_bounds__(256) void k_gemm_bf16(
    const bf16_t* __restrict__ A, const bf16_t* __restrict__ B,
    float* __restrict__ C, int M, int N, int K) {
  __shared__ __align__(32) bf16_t As[2][BM][BK];   // 2 x 8 KB
  __shared__ __align__(32) bf16_t Bs[2][BK][BN];   // 2 x 8 KB
  int tid  = threadIdx.x;
  int wave = tid >> 5, lane = tid & 31;
  int half = lane >> 4, r16 = lane & 15;
  int rowBase = blockIdx.y * BM;
  int colBase = blockIdx.x * BN;

  auto stage = [&](int buf, int k0) {
#pragma unroll
    for (int i = 0; i < 2; ++i) {            // A: 128x32 = 512 x 16B chunks
      int chunk = tid + i * 256;
      int ar = chunk >> 2, ac = (chunk & 3) * 8;
      STAGE16B(&A[(size_t)(rowBase + ar) * K + k0 + ac], &As[buf][ar][ac]);
    }
#pragma unroll
    for (int i = 0; i < 2; ++i) {            // B: 32x128 = 512 x 16B chunks
      int chunk = tid + i * 256;
      int br = chunk >> 4, bc = (chunk & 15) * 8;
      STAGE16B(&B[(size_t)(k0 + br) * N + colBase + bc], &Bs[buf][br][bc]);
    }
  };

  v8f z = {0.f, 0.f, 0.f, 0.f, 0.f, 0.f, 0.f, 0.f};
  v8f acc[BN / 16];
#pragma unroll
  for (int n = 0; n < BN / 16; ++n) acc[n] = z;

  stage(0, 0);
  async_wait0();
  __syncthreads();

  int nk = K / BK;
  for (int t = 0; t < nk; ++t) {
    int buf = t & 1;
    if (t + 1 < nk) stage(buf ^ 1, (t + 1) * BK);  // overlap with compute

    int arow = wave * 16 + r16;
    v16bf a;
    const bf16_t* ap = &As[buf][arow][half * 8];
    ((uint4*)&a)[0] = *(const uint4*)(ap);
    ((uint4*)&a)[1] = *(const uint4*)(ap + 16);
#pragma unroll
    for (int n = 0; n < BN / 16; ++n) {
      v16bf b;
#pragma unroll
      for (int s = 0; s < 16; ++s)
        b[s] = Bs[buf][half * 8 + (s & 7) + (s >> 3) * 16][n * 16 + r16];
      acc[n] = __builtin_amdgcn_wmma_f32_16x16x32_bf16(
          false, a, false, b, (short)0, acc[n], false, false);
    }
    async_wait0();
    __syncthreads();
  }

#pragma unroll
  for (int n = 0; n < BN / 16; ++n)
#pragma unroll
    for (int r = 0; r < 8; ++r) {
      int row = rowBase + wave * 16 + r + 8 * half;
      int col = colBase + n * 16 + r16;
      C[(size_t)row * N + col] = acc[n][r];
    }
}

// --------------------------------------------------------------- attention
// One workgroup = 64 queries x 1 head (128 thr, 4 waves), flash-style online
// softmax over causal 64-key blocks. K comes pre-transposed: Kt [NKV*HD][T].
// Q: bf16 [T][4096], Kt: bf16 [1024][T], V: bf16 [T][1024], O: bf16 [T][4096]
__global__ __launch_bounds__(128) void k_attn(
    const bf16_t* __restrict__ Q, const bf16_t* __restrict__ Kt,
    const bf16_t* __restrict__ V, bf16_t* __restrict__ O, int T) {
  __shared__ __align__(32) bf16_t Qs[64][HD];       // 16 KB
  __shared__ __align__(32) bf16_t KsT[2][HD][64];   // 2 x 16 KB  ([d][key])
  __shared__ __align__(32) bf16_t Vs[2][64][HD];    // 2 x 16 KB
  __shared__ float Ss[64][64];                      // 16 KB
  __shared__ __align__(32) bf16_t Ps[64][64];       //  8 KB
  __shared__ float mrow[64], lrow[64], alph[64];

  int tid  = threadIdx.x;
  int wave = tid >> 5, lane = tid & 31;
  int half = lane >> 4, r16 = lane & 15;
  int h    = blockIdx.y;
  int kvh  = h >> 2;               // GQA: 4 query heads per kv head
  int qBase = blockIdx.x * 64;
  const float scale = 0.08838834764831843f;  // 1/sqrt(128)

  auto stageKV = [&](int buf, int kb) {
    int kBase = kb * 64;
#pragma unroll
    for (int i = 0; i < 8; ++i) {            // K^T tile: 128d x 64k
      int chunk = tid + i * 128;             // 0..1023
      int d = chunk >> 3;
      int k8 = (chunk & 7) * 8;
      STAGE16B(&Kt[((size_t)(kvh * HD + d)) * T + kBase + k8],
               &KsT[buf][d][k8]);
    }
#pragma unroll
    for (int i = 0; i < 8; ++i) {            // V tile: 64k x 128d
      int chunk = tid + i * 128;
      int key = chunk >> 4;
      int d8 = (chunk & 15) * 8;
      STAGE16B(&V[(size_t)(kBase + key) * KVD + kvh * HD + d8],
               &Vs[buf][key][d8]);
    }
  };

  // stage Q tile + first K/V block
#pragma unroll
  for (int i = 0; i < 8; ++i) {
    int chunk = tid + i * 128;
    int row = chunk >> 4;
    int c8 = (chunk & 15) * 8;
    STAGE16B(&Q[(size_t)(qBase + row) * DMODEL + h * HD + c8], &Qs[row][c8]);
  }
  stageKV(0, 0);
  if (tid < 64) { mrow[tid] = -1e30f; lrow[tid] = 0.f; }
  async_wait0();
  __syncthreads();

  v8f z = {0.f, 0.f, 0.f, 0.f, 0.f, 0.f, 0.f, 0.f};
  v8f oacc[8];
#pragma unroll
  for (int n = 0; n < 8; ++n) oacc[n] = z;

  int nkb = blockIdx.x + 1;  // causal: only key blocks <= query block
  for (int kb = 0; kb < nkb; ++kb) {
    int kBase = kb * 64;
    int buf = kb & 1;
    if (kb + 1 < nkb) stageKV(buf ^ 1, kb + 1);  // overlap with compute

    // S = Q . K^T  (16 queries/wave x 64 keys, accumulate over d=128)
    v8f sacc[4];
#pragma unroll
    for (int n = 0; n < 4; ++n) sacc[n] = z;
#pragma unroll
    for (int c = 0; c < 4; ++c) {
      v16bf a;
      const bf16_t* ap = &Qs[wave * 16 + r16][c * 32 + half * 8];
      ((uint4*)&a)[0] = *(const uint4*)(ap);
      ((uint4*)&a)[1] = *(const uint4*)(ap + 16);
#pragma unroll
      for (int n = 0; n < 4; ++n) {
        v16bf b;
#pragma unroll
        for (int s = 0; s < 16; ++s)
          b[s] = KsT[buf][c * 32 + half * 8 + (s & 7) + (s >> 3) * 16][n * 16 + r16];
        sacc[n] = __builtin_amdgcn_wmma_f32_16x16x32_bf16(
            false, a, false, b, (short)0, sacc[n], false, false);
      }
    }
    // scaled + causal-masked S -> LDS
#pragma unroll
    for (int n = 0; n < 4; ++n)
#pragma unroll
      for (int r = 0; r < 8; ++r) {
        int row = wave * 16 + r + 8 * half;
        int col = n * 16 + r16;
        float sv = sacc[n][r] * scale;
        if (kBase + col > qBase + row) sv = -1e30f;
        Ss[row][col] = sv;
      }
    __syncthreads();

    // online softmax: thread t owns query row t
    if (tid < 64) {
      float mx = mrow[tid];
      for (int c2 = 0; c2 < 64; ++c2) mx = fmaxf(mx, Ss[tid][c2]);
      float al = __expf(mrow[tid] - mx);
      float sum = 0.f;
      for (int c2 = 0; c2 < 64; ++c2) {
        float p = __expf(Ss[tid][c2] - mx);
        sum += p;
        Ps[tid][c2] = f2bf(p);
      }
      mrow[tid] = mx;
      lrow[tid] = lrow[tid] * al + sum;
      alph[tid] = al;
    }
    __syncthreads();

    // rescale O accumulators
#pragma unroll
    for (int r = 0; r < 8; ++r) {
      float al = alph[wave * 16 + r + 8 * half];
#pragma unroll
      for (int n = 0; n < 8; ++n) oacc[n][r] *= al;
    }
    // O += P . V
#pragma unroll
    for (int c = 0; c < 2; ++c) {
      v16bf a;
      const bf16_t* ap = &Ps[wave * 16 + r16][c * 32 + half * 8];
      ((uint4*)&a)[0] = *(const uint4*)(ap);
      ((uint4*)&a)[1] = *(const uint4*)(ap + 16);
#pragma unroll
      for (int n = 0; n < 8; ++n) {
        v16bf b;
#pragma unroll
        for (int s = 0; s < 16; ++s)
          b[s] = Vs[buf][c * 32 + half * 8 + (s & 7) + (s >> 3) * 16][n * 16 + r16];
        oacc[n] = __builtin_amdgcn_wmma_f32_16x16x32_bf16(
            false, a, false, b, (short)0, oacc[n], false, false);
      }
    }
    async_wait0();
    __syncthreads();
  }

  // epilogue: O / l
#pragma unroll
  for (int n = 0; n < 8; ++n)
#pragma unroll
    for (int r = 0; r < 8; ++r) {
      int row = wave * 16 + r + 8 * half;
      int col = n * 16 + r16;
      float ov = oacc[n][r] / lrow[row];
      O[(size_t)(qBase + row) * DMODEL + h * HD + col] = f2bf(ov);
    }
}

// ------------------------------------------------------------------- host
extern "C" void kernel_launch(void* const* d_in, const int* in_sizes, int n_in,
                              void* d_out, int out_size, void* d_ws,
                              size_t ws_size, hipStream_t stream) {
  (void)n_in; (void)out_size; (void)ws_size;
  const float* x  = (const float*)d_in[0];
  const float* wq = (const float*)d_in[1];
  const float* wk = (const float*)d_in[2];
  const float* wv = (const float*)d_in[3];
  const float* wo = (const float*)d_in[4];
  int T = in_sizes[0] / DMODEL;  // 2048

  char* ws = (char*)d_ws;
  size_t off = 0;
  auto alloc = [&](size_t bytes) -> void* {
    void* p = ws + off;
    off += (bytes + 255) & ~(size_t)255;
    return p;
  };
  bf16_t* xb  = (bf16_t*)alloc((size_t)T * DMODEL * 2);
  bf16_t* wqb = (bf16_t*)alloc((size_t)DMODEL * DMODEL * 2);
  bf16_t* wkb = (bf16_t*)alloc((size_t)DMODEL * KVD * 2);
  bf16_t* wvb = (bf16_t*)alloc((size_t)DMODEL * KVD * 2);
  bf16_t* wob = (bf16_t*)alloc((size_t)DMODEL * DMODEL * 2);
  float*  qf  = (float*)alloc((size_t)T * DMODEL * 4);
  float*  kf  = (float*)alloc((size_t)T * KVD * 4);
  float*  vf  = (float*)alloc((size_t)T * KVD * 4);
  bf16_t* qbb = (bf16_t*)alloc((size_t)T * DMODEL * 2);
  bf16_t* ktb = (bf16_t*)alloc((size_t)T * KVD * 2);   // transposed K
  bf16_t* vbb = (bf16_t*)alloc((size_t)T * KVD * 2);
  bf16_t* aob = (bf16_t*)alloc((size_t)T * DMODEL * 2);

  auto cvt = [&](const float* in, bf16_t* out, long n) {
    int blocks = (int)((n + 256 * 8 - 1) / (256 * 8));
    k_f32_to_bf16<<<blocks, 256, 0, stream>>>(in, out, n);
  };
  cvt(x,  xb,  (long)T * DMODEL);
  cvt(wq, wqb, (long)DMODEL * DMODEL);
  cvt(wk, wkb, (long)DMODEL * KVD);
  cvt(wv, wvb, (long)DMODEL * KVD);
  cvt(wo, wob, (long)DMODEL * DMODEL);

  dim3 blk(256);
  k_gemm_bf16<<<dim3(DMODEL / BN, T / BM), blk, 0, stream>>>(
      xb, wqb, qf, T, DMODEL, DMODEL);
  k_gemm_bf16<<<dim3(KVD / BN, T / BM), blk, 0, stream>>>(
      xb, wkb, kf, T, KVD, DMODEL);
  k_gemm_bf16<<<dim3(KVD / BN, T / BM), blk, 0, stream>>>(
      xb, wvb, vf, T, KVD, DMODEL);

  {
    int nq = T * NH * (HD / 2);
    k_rope_to_bf16<<<(nq + 255) / 256, 256, 0, stream>>>(qf, qbb, T, NH);
    int nk = T * NKV * (HD / 2);
    k_rope_kt<<<(nk + 255) / 256, 256, 0, stream>>>(kf, ktb, T);
  }
  cvt(vf, vbb, (long)T * KVD);

  k_attn<<<dim3(T / 64, NH), dim3(128), 0, stream>>>(qbb, ktb, vbb, aob, T);

  k_gemm_bf16<<<dim3(DMODEL / BN, T / BM), blk, 0, stream>>>(
      aob, wob, (float*)d_out, T, DMODEL, DMODEL);
}
